// GraphConvolution_9809705304175
// MI455X (gfx1250) — matrix operand order repaired and checked by reading, not compiled, not run on hardware
//
#include <hip/hip_runtime.h>

typedef __attribute__((ext_vector_type(16))) _Float16 v16h;
typedef __attribute__((ext_vector_type(8)))  _Float16 v8h;
typedef __attribute__((ext_vector_type(8)))  float    v8f;

#define N_NODES 8192
#define IN_F    1024
#define OUT_F   512
#define N_EDGES 262144

// workspace layout (bytes)
#define XH_OFF   (0ull)          // x in f16:      8192*1024*2 = 16 MiB
#define WH_OFF   (16ull << 20)   // W in f16:      512*1024*2  =  1 MiB
#define HID_OFF  (17ull << 20)   // hidden f32:    8192*512*4  = 16 MiB
#define BM_OFF   (33ull << 20)   // adjacency bitmap: 8192*8192 bits = 8 MiB
#define BM_WORDS (N_NODES * (N_NODES / 32))   // 2,097,152 u32

union AFrag { v16h v; v8h h[2]; };
union BFrag { v16h v; v8h h[2]; };

// ---------------------------------------------------------------------------
// Phase 0: convert x and W to f16 (feeds the WMMA f16 path)
// ---------------------------------------------------------------------------
__global__ void convert_f16_kernel(const float* __restrict__ x,
                                   const float* __restrict__ W,
                                   _Float16* __restrict__ xh,
                                   _Float16* __restrict__ wh) {
    const unsigned XN = N_NODES * IN_F;          // 8,388,608
    const unsigned WN = OUT_F * IN_F;            //   524,288
    unsigned i = blockIdx.x * blockDim.x + threadIdx.x;
    if (i < XN) {
        xh[i] = (_Float16)x[i];
    } else if (i < XN + WN) {
        unsigned j = i - XN;
        wh[j] = (_Float16)W[j];
    }
}

// ---------------------------------------------------------------------------
// Phase 1: zero the adjacency bitmap
// ---------------------------------------------------------------------------
__global__ void bitmap_zero_kernel(unsigned* __restrict__ bm) {
    unsigned i = blockIdx.x * blockDim.x + threadIdx.x;
    if (i < BM_WORDS) bm[i] = 0u;
}

// ---------------------------------------------------------------------------
// Phase 2: scatter edges into bitmap (atomicOr dedupes duplicate edges,
// matching the reference's adj.at[...].set(1.0) semantics)
// ---------------------------------------------------------------------------
__global__ void edge_bitmap_kernel(const long long* __restrict__ src,
                                   const long long* __restrict__ dst,
                                   unsigned* __restrict__ bm) {
    unsigned e = blockIdx.x * blockDim.x + threadIdx.x;
    if (e < N_EDGES) {
        unsigned s = (unsigned)src[e];
        unsigned d = (unsigned)dst[e];
        unsigned bit = s * (unsigned)N_NODES + d;   // row = src, col = dst
        atomicOr(&bm[bit >> 5], 1u << (bit & 31u));
    }
}

// ---------------------------------------------------------------------------
// Fragment loaders. Contiguous K order used here, same for A and B (wave32):
//  A 16x32 f16: lane L<16 -> row M=L,   K in {0..15}
//               lane L>=16 -> row M=L-16, K in {16..31}
//    => two contiguous 16B loads per lane at base = 16*halfSel and base+8.
//  B 32x16 f16: col N = lane%16, K half = 16*halfSel, 16 contiguous f16.
// ---------------------------------------------------------------------------
__device__ __forceinline__ void load_a(AFrag& a, const _Float16* arow, unsigned k0) {
    a.h[0] = *(const v8h*)(arow + k0);
    a.h[1] = *(const v8h*)(arow + k0 + 8u);
}
__device__ __forceinline__ void load_b4(BFrag b[4], const _Float16* brow, unsigned k0) {
    #pragma unroll
    for (unsigned t = 0; t < 4; ++t) {
        const _Float16* p = brow + (size_t)t * 16u * IN_F + k0;
        b[t].h[0] = *(const v8h*)(p);
        b[t].h[1] = *(const v8h*)(p + 8u);
    }
}

// ---------------------------------------------------------------------------
// Phase 3: hidden = x @ W^T + b via v_wmma_f32_16x16x32_f16.
// Each wave computes a 16x64 output strip (1 A-frag reused over 4 B-frags).
// Register double-buffering: fragments for K-step k+32 are issued before the
// WMMAs of step k, so loads overlap matrix ops instead of s_wait_loadcnt 0.
// grid: 512 blocks x 256 threads = 4096 waves = 512 M-tiles x 8 strip-cols.
// ---------------------------------------------------------------------------
__global__ void __launch_bounds__(256)
gemm_wmma_kernel(const _Float16* __restrict__ xh,
                 const _Float16* __restrict__ wh,
                 const float* __restrict__ bias,
                 float* __restrict__ hidden) {
    const unsigned lane  = threadIdx.x & 31u;
    const unsigned wave  = threadIdx.x >> 5;
    const unsigned w     = blockIdx.x * 8u + wave;   // global wave id [0,4096)
    const unsigned mTile = w >> 3;                   // [0,512)
    const unsigned n0    = (w & 7u) * 64u;           // strip of 4 N-tiles

    const unsigned m0      = mTile * 16u;
    const unsigned colBase = lane & 15u;
    const unsigned halfSel = (lane >> 4) & 1u;

    v8f acc[4];
    acc[0] = (v8f)0.0f; acc[1] = (v8f)0.0f; acc[2] = (v8f)0.0f; acc[3] = (v8f)0.0f;

    const _Float16* arow = xh + (size_t)(m0 + colBase) * IN_F + halfSel * 16u;
    const _Float16* brow = wh + (size_t)(n0 + colBase) * IN_F + halfSel * 16u;

    AFrag a0, a1;
    BFrag b0[4], b1[4];

    // prologue: fragments for k = 0
    load_a(a0, arow, 0u);
    load_b4(b0, brow, 0u);

    for (unsigned k0 = 0; k0 < IN_F; k0 += 64u) {
        // prefetch k0+32 while computing k0
        load_a(a1, arow, k0 + 32u);
        load_b4(b1, brow, k0 + 32u);
        #pragma unroll
        for (unsigned t = 0; t < 4; ++t)
            acc[t] = __builtin_amdgcn_wmma_f32_16x16x32_f16(
                         false, a0.v, false, b0[t].v, (short)0, acc[t], false, false);

        // prefetch k0+64 while computing k0+32
        if (k0 + 64u < IN_F) {
            load_a(a0, arow, k0 + 64u);
            load_b4(b0, brow, k0 + 64u);
        }
        #pragma unroll
        for (unsigned t = 0; t < 4; ++t)
            acc[t] = __builtin_amdgcn_wmma_f32_16x16x32_f16(
                         false, a1.v, false, b1[t].v, (short)0, acc[t], false, false);
    }

    // C/D layout: VGPR r -> row m0 + r + 8*halfSel, col = tile base + lane%16
    const unsigned rowOff = halfSel * 8u;
    #pragma unroll
    for (unsigned t = 0; t < 4; ++t) {
        const unsigned col = n0 + t * 16u + colBase;
        const float bv = bias[col];
        #pragma unroll
        for (unsigned r = 0; r < 8; ++r) {
            hidden[(size_t)(m0 + rowOff + r) * OUT_F + col] = acc[t][r] + bv;
        }
    }
}

// ---------------------------------------------------------------------------
// Phase 4: out[i] = relu( sum_{j: adj[i][j]} hidden[j] ).
// One block per output row; bitmap row staged in LDS; all 256 threads walk
// the same (uniform) bit pattern, each owning 2 of the 512 columns ->
// fully coalesced 2KB gathers of hidden rows (resident in 192MB L2).
// ---------------------------------------------------------------------------
__global__ void __launch_bounds__(256)
aggregate_kernel(const unsigned* __restrict__ bm,
                 const float* __restrict__ hidden,
                 float* __restrict__ out) {
    __shared__ unsigned words[N_NODES / 32];   // 256 u32 = this row's adjacency
    const unsigned row = blockIdx.x;
    const unsigned tid = threadIdx.x;

    words[tid] = bm[(size_t)row * (N_NODES / 32) + tid];
    __syncthreads();

    float acc0 = 0.0f, acc1 = 0.0f;

    for (unsigned wIdx = 0; wIdx < N_NODES / 32; ++wIdx) {
        unsigned bits = words[wIdx];           // uniform across the block
        while (bits) {
            unsigned j = (unsigned)__builtin_ctz(bits);
            bits &= bits - 1u;
            const float* h = hidden + (size_t)(wIdx * 32u + j) * OUT_F;
            float2 hv = ((const float2*)h)[tid];
            acc0 += hv.x;
            acc1 += hv.y;
        }
    }

    float2 res;
    res.x = fmaxf(acc0, 0.0f);
    res.y = fmaxf(acc1, 0.0f);
    ((float2*)(out + (size_t)row * OUT_F))[tid] = res;
}

// ---------------------------------------------------------------------------
extern "C" void kernel_launch(void* const* d_in, const int* in_sizes, int n_in,
                              void* d_out, int out_size, void* d_ws, size_t ws_size,
                              hipStream_t stream) {
    const float*     x  = (const float*)d_in[0];
    const float*     W  = (const float*)d_in[1];
    const float*     b  = (const float*)d_in[2];
    const long long* ei = (const long long*)d_in[3];   // [2, N_EDGES] int64
    float* out = (float*)d_out;

    char* ws = (char*)d_ws;
    _Float16* xh     = (_Float16*)(ws + XH_OFF);
    _Float16* wh     = (_Float16*)(ws + WH_OFF);
    float*    hidden = (float*)   (ws + HID_OFF);
    unsigned* bm     = (unsigned*)(ws + BM_OFF);

    // total f16 conversions: 8192*1024 + 512*1024 = 8,912,896 = 34816 * 256
    convert_f16_kernel<<<34816, 256, 0, stream>>>(x, W, xh, wh);
    bitmap_zero_kernel<<<BM_WORDS / 256, 256, 0, stream>>>(bm);
    edge_bitmap_kernel<<<N_EDGES / 256, 256, 0, stream>>>(ei, ei + N_EDGES, bm);
    gemm_wmma_kernel<<<512, 256, 0, stream>>>(xh, wh, b, hidden);
    aggregate_kernel<<<N_NODES, 256, 0, stream>>>(bm, hidden, out);
}
